// StackedGaussianMixtureModel_66194035965993
// MI455X (gfx1250) — matrix operand order repaired and checked by reading, not compiled, not run on hardware
//
#include <hip/hip_runtime.h>
#include <hip/hip_bf16.h>

// ---------------------------------------------------------------------------
// Stacked GMM M-step for MI455X (gfx1250, wave32, WMMA).
//   out = [ pi(32) | mus(32x32) | covs(32x32x32) ]  (f32, 33824 elements)
// Heavy term m2[c] = X^T diag(w_c) X via v_wmma_f32_16x16x32_bf16
// (bf16 inputs, f32 accumulate); mus/denom/pi stay exact f32.
// 512-thread blocks (16 wave32s), 2 components/wave -> 8 accumulator tiles
// (64 VGPRs) per wave: keeps total VGPRs < 256 (no s_set_vgpr_msb churn),
// higher occupancy to hide ds_load latency + WMMA hazard windows.
// ---------------------------------------------------------------------------

#define NPTS  500000
#define KTOP  16
#define DDIM  32
#define CCOMP 32        // K*S
#define CHUNK 128       // points staged per block iteration
#define TPB   512       // 16 wave32s
#define EPSV  1e-10f

typedef __attribute__((ext_vector_type(16))) __bf16 v16bf;
typedef __attribute__((ext_vector_type(16))) float  v16f;
typedef __attribute__((ext_vector_type(8)))  float  v8f;

// d_ws layout (floats):
//   [0      , 32768) : m2 accumulator      [c][i][j]
//   [32768  , 33792) : mus numerator       [c][d]
//   [33792  , 33824) : denom               [c]
//   [33824  , 33856) : pi (ri column sums) [c]
#define WS_M2   0
#define WS_MU   32768
#define WS_DEN  (32768 + 1024)
#define WS_PI   (32768 + 1024 + 32)
#define WS_TOT  (32768 + 1024 + 32 + 32)

__global__ void gmm_zero_ws(float* __restrict__ ws) {
  int i = blockIdx.x * 256 + threadIdx.x;
  if (i < WS_TOT) ws[i] = 0.f;
}

__global__ void __launch_bounds__(TPB, 1) gmm_mstep_main(
    const float* __restrict__ X, const float* __restrict__ R,
    const float* __restrict__ RI, float* __restrict__ ws)
{
  // LDS staging in exact WMMA operand layouts (ISA 7.12.2):
  //  A (16x32 bf16): lane = halfA*16 + (col&15); bf16 slot j covers K per
  //    half-lane split {0-7,16-23}/{8-15,24-31}. Stored as f32, scaled by w
  //    and converted per-component in registers (f32 mul -> cvt_pk_bf16).
  //  B (32x16 bf16): lane = halfB*16 + (col&15); same K split as A.
  __shared__ __attribute__((aligned(64))) float  sA[2][4][32][16]; // 16 KB
  __shared__ __attribute__((aligned(16))) __bf16 sB[2][4][32][16]; //  8 KB
  __shared__ __attribute__((aligned(64))) float  sW[CCOMP][4][2][16]; // 16 KB
  __shared__ float sMu[CCOMP][DDIM];   // 4 KB  block-partial mus numerator
  __shared__ float sDen[CCOMP];
  __shared__ float sPi[CCOMP];
  __shared__ int   sZ[CHUNK];

  const int tid  = threadIdx.x;
  const int lane = tid & 31;
  const int wave = tid >> 5;          // 0..15
  const int c0   = wave * 2;          // 2 components per wave

  for (int i = tid; i < CCOMP * DDIM; i += TPB) ((float*)sMu)[i] = 0.f;
  if (tid < CCOMP) { sDen[tid] = 0.f; sPi[tid] = 0.f; }

  v8f acc[2][2][2];                    // 8 tiles x 8 VGPRs, f32 accumulate
  #pragma unroll
  for (int a = 0; a < 2; a++)
    #pragma unroll
    for (int b = 0; b < 2; b++)
      #pragma unroll
      for (int c = 0; c < 2; c++) acc[a][b][c] = v8f{};

  float piAcc[8];
  #pragma unroll
  for (int i = 0; i < 8; i++) piAcc[i] = 0.f;

  // staging geometry: 4 threads per point (one 8-dim slice each)
  const int p    = tid >> 2;          // local point 0..127
  const int h    = tid & 3;           // dim quarter
  const int ksl  = p >> 5;            // K-step 0..3
  const int Kl   = p & 31;            // K within step
  const int halfA = (Kl >> 3) & 1;                 // A-operand half-lane
  const int jA    = (Kl & 7) | ((Kl >> 4) << 3);   // A-operand bf16 slot
  const int halfB = (Kl >> 3) & 1;                 // B-operand half-lane
  const int jB    = (Kl & 7) | ((Kl >> 4) << 3);   // B-operand bf16 slot

  const long long NC = (NPTS + CHUNK - 1) / CHUNK;
  __syncthreads();

  for (long long ch = blockIdx.x; ch < NC; ch += (long long)gridDim.x) {
    const long long gp = ch * CHUNK + p;
    const bool valid = gp < NPTS;

    // ---------- phase A: argmax z, stage X into A/B layouts, pi ----------
    if (h == 0) {
      int z = 0; float best = -3.4e38f;
      if (valid) {
        const float* rr = R + gp * KTOP;
        #pragma unroll
        for (int k = 0; k < KTOP; k++) { float v = rr[k]; if (v > best) { best = v; z = k; } }
      }
      sZ[p] = z;
    }

    float x[8];
    if (valid) {
      const float4* xv = reinterpret_cast<const float4*>(X + gp * DDIM + h * 8);
      #pragma unroll
      for (int q = 0; q < 2; q++) {
        float4 t = xv[q];
        x[4*q+0] = t.x; x[4*q+1] = t.y; x[4*q+2] = t.z; x[4*q+3] = t.w;
      }
      const long long np = gp + (long long)gridDim.x * CHUNK;
      if (np < NPTS)  // speculative prefetch of next chunk -> global_prefetch_b8
        __builtin_prefetch(X + np * DDIM + h * 8, 0, 1);
    } else {
      #pragma unroll
      for (int i = 0; i < 8; i++) x[i] = 0.f;
    }

    #pragma unroll
    for (int i = 0; i < 8; i++) {
      const int d = h * 8 + i;
      sA[d >> 4][ksl][halfA * 16 + (d & 15)][jA] = x[i];
      sB[d >> 4][ksl][halfB * 16 + (d & 15)][jB] = (__bf16)x[i];
    }

    if (valid) {  // ungated ri column sums (pi); fixed columns per thread
      const float* rrow = RI + gp * CCOMP + h * 8;
      #pragma unroll
      for (int i = 0; i < 8; i++) piAcc[i] += rrow[i];
    }

    __syncthreads();

    // ---------- phase B: gated weights (2-of-32 sparse), mus/denom -------
    const int z = sZ[p];
    float w0 = 0.f, w1 = 0.f;
    if (valid) { w0 = RI[gp * CCOMP + 2 * z]; w1 = RI[gp * CCOMP + 2 * z + 1]; }
    // each of the 4 point-threads owns an 8-component octet of sW; 2z and
    // 2z+1 always fall in the same octet (octets are 8-aligned), so this is
    // race-free with no extra barrier.
    {
      const int cbase = h * 8;
      #pragma unroll
      for (int q = 0; q < 8; q++) {
        const int c = cbase + q;
        float wv = 0.f;
        if (c == 2 * z)     wv = w0;
        if (c == 2 * z + 1) wv = w1;
        sW[c][ksl][halfA][jA] = wv;
      }
    }
    if (h == 0) {
      unsafeAtomicAdd(&sDen[2 * z],     w0);   // ds_add_f32
      unsafeAtomicAdd(&sDen[2 * z + 1], w1);
    }
    #pragma unroll
    for (int i = 0; i < 8; i++) {
      const int d = h * 8 + i;
      unsafeAtomicAdd(&sMu[2 * z    ][d], w0 * x[i]);
      unsafeAtomicAdd(&sMu[2 * z + 1][d], w1 * x[i]);
    }

    __syncthreads();

    // ---------- compute: 32 v_wmma_f32_16x16x32_bf16 per wave/chunk ------
    #pragma unroll
    for (int ks = 0; ks < 4; ks++) {
      const v16bf b0 = *(const v16bf*)&sB[0][ks][lane][0];
      const v16bf b1 = *(const v16bf*)&sB[1][ks][lane][0];
      const v16f xa0 = *(const v16f*)&sA[0][ks][lane][0];
      const v16f xa1 = *(const v16f*)&sA[1][ks][lane][0];
      #pragma unroll
      for (int cc = 0; cc < 2; cc++) {
        const v16f wv = *(const v16f*)&sW[c0 + cc][ks][lane >> 4][0];
        const v16bf a0 = __builtin_convertvector(xa0 * wv, v16bf); // f32 scale -> bf16
        const v16bf a1 = __builtin_convertvector(xa1 * wv, v16bf);
        acc[cc][0][0] = __builtin_amdgcn_wmma_f32_16x16x32_bf16(false, a0, false, b0, (short)0, acc[cc][0][0], false, false);
        acc[cc][0][1] = __builtin_amdgcn_wmma_f32_16x16x32_bf16(false, a0, false, b1, (short)0, acc[cc][0][1], false, false);
        acc[cc][1][0] = __builtin_amdgcn_wmma_f32_16x16x32_bf16(false, a1, false, b0, (short)0, acc[cc][1][0], false, false);
        acc[cc][1][1] = __builtin_amdgcn_wmma_f32_16x16x32_bf16(false, a1, false, b1, (short)0, acc[cc][1][1], false, false);
      }
    }
    __syncthreads();   // LDS reused next chunk
  }

  // ---------- flush block partials ----------
  #pragma unroll
  for (int i = 0; i < 8; i++) unsafeAtomicAdd(&sPi[h * 8 + i], piAcc[i]);
  __syncthreads();

  float* m2ws  = ws + WS_M2;
  float* muws  = ws + WS_MU;
  float* denws = ws + WS_DEN;
  float* piws  = ws + WS_PI;

  // C/D layout: VGPR v holds M = v (lanes 0-15) / 8+v (lanes 16-31), N = lane&15
  const int Ncol = lane & 15;
  const int Mb   = (lane >> 4) * 8;
  #pragma unroll
  for (int cc = 0; cc < 2; cc++)
    #pragma unroll
    for (int mt = 0; mt < 2; mt++)
      #pragma unroll
      for (int nt = 0; nt < 2; nt++)
        #pragma unroll
        for (int v = 0; v < 8; v++) {
          const int c = c0 + cc;
          const int M = mt * 16 + Mb + v;
          unsafeAtomicAdd(&m2ws[(c * 32 + M) * 32 + nt * 16 + Ncol],
                          acc[cc][mt][nt][v]);   // global_atomic_add_f32
        }

  for (int i = tid; i < CCOMP * DDIM; i += TPB)
    unsafeAtomicAdd(&muws[i], ((const float*)sMu)[i]);
  if (tid < CCOMP) {
    unsafeAtomicAdd(&denws[tid], sDen[tid]);
    unsafeAtomicAdd(&piws[tid],  sPi[tid]);
  }
}

__global__ void gmm_finalize(const float* __restrict__ ws, float* __restrict__ out) {
  const int idx = blockIdx.x * 256 + threadIdx.x;
  const float* m2ws  = ws + WS_M2;
  const float* muws  = ws + WS_MU;
  const float* denws = ws + WS_DEN;
  const float* piws  = ws + WS_PI;
  if (idx < 32) {
    out[idx] = piws[idx] * (1.0f / (float)NPTS);
  } else if (idx < 32 + 1024) {
    const int t = idx - 32; const int c = t >> 5;
    out[idx] = muws[t] / fmaxf(denws[c], EPSV);
  } else if (idx < 32 + 1024 + 32768) {
    const int t = idx - 1056;
    const int c = t >> 10; const int i = (t >> 5) & 31; const int j = t & 31;
    const float s  = fmaxf(denws[c], EPSV);
    const float mi = muws[c * 32 + i] / s;
    const float mj = muws[c * 32 + j] / s;
    out[idx] = m2ws[t] / s - mi * mj;   // E[xx^T] - mu mu^T
  }
}

extern "C" void kernel_launch(void* const* d_in, const int* in_sizes, int n_in,
                              void* d_out, int out_size, void* d_ws, size_t ws_size,
                              hipStream_t stream) {
  const float* X  = (const float*)d_in[0];   // [N,32]
  const float* R  = (const float*)d_in[1];   // [N,16]
  const float* RI = (const float*)d_in[2];   // [N,32]
  float* ws  = (float*)d_ws;
  float* out = (float*)d_out;                // 33824 floats

  gmm_zero_ws<<<(WS_TOT + 255) / 256, 256, 0, stream>>>(ws);
  gmm_mstep_main<<<1024, TPB, 0, stream>>>(X, R, RI, ws);
  gmm_finalize<<<(33824 + 255) / 256, 256, 0, stream>>>(ws, out);
}
